// GraphConvolution_82282983457294
// MI455X (gfx1250) — compile-verified, matrix operand order checked by reading, not run on hardware
//
#include <hip/hip_runtime.h>

typedef __attribute__((ext_vector_type(16))) __bf16 v16bf;
typedef __attribute__((ext_vector_type(8)))  float  v8f;

#define NROWS 8192
#define IN_F  512
#define OUT_F 256
#define NEG_BIG (-1.0e30f)

__device__ __forceinline__ unsigned short f2bf(float f) {
  union { float f; unsigned int u; } v; v.f = f;
  unsigned int u = v.u;
  return (unsigned short)((u + 0x7FFFu + ((u >> 16) & 1u)) >> 16);
}

// ---------------- Kernel 0: convert x -> bf16, W -> W^T bf16 -----------------
__global__ void gat_convert(const float* __restrict__ x, const float* __restrict__ W,
                            unsigned short* __restrict__ x_bf,
                            unsigned short* __restrict__ wt_bf) {
  size_t gid = (size_t)blockIdx.x * blockDim.x + threadIdx.x;
  if (gid < (size_t)NROWS * IN_F) x_bf[gid] = f2bf(x[gid]);
  if (gid < (size_t)IN_F * OUT_F) {
    int c = (int)(gid >> 9);        // 0..255
    int k = (int)(gid & 511);       // 0..511
    wt_bf[(size_t)c * IN_F + k] = f2bf(W[(size_t)k * OUT_F + c]);
  }
}

// ---------------- Kernel 1: h = x @ W ; g = h*a/8 ; hT -----------------------
__global__ __launch_bounds__(128) void gat_hgemm(
    const unsigned short* __restrict__ x_bf, const unsigned short* __restrict__ wt_bf,
    const float* __restrict__ a_vec,
    unsigned short* __restrict__ h_bf, unsigned short* __restrict__ g_bf,
    unsigned short* __restrict__ hT_bf) {
  const int lane = threadIdx.x & 31;
  const int lm = lane & 15;
  const int hh = lane >> 4;
  const int waveId = blockIdx.x * (blockDim.x >> 5) + (threadIdx.x >> 5);
  const int rbase = (waveId >> 4) << 4;   // 512 row tiles
  const int cbase = (waveId & 15) << 4;   // 16 col tiles

  v8f c = 0.f;
  const unsigned short* xr = x_bf + (size_t)(rbase + lm) * IN_F + 16 * hh;
  const unsigned short* wr = wt_bf + (size_t)(cbase + lm) * IN_F + 16 * hh;
#pragma unroll
  for (int k0 = 0; k0 < IN_F; k0 += 32) {
    v16bf a = *(const v16bf*)(xr + k0);
    v16bf b = *(const v16bf*)(wr + k0);
    c = __builtin_amdgcn_wmma_f32_16x16x32_bf16(false, a, false, b, (short)0, c, false, false);
  }
  const float av = a_vec[cbase + lm] * 0.125f;  // fold 1/SCORE_DIV into g
#pragma unroll
  for (int r = 0; r < 8; ++r) {
    const int row = rbase + r + 8 * hh;
    const int col = cbase + lm;
    unsigned short hb = f2bf(c[r]);
    h_bf[(size_t)row * OUT_F + col] = hb;
    g_bf[(size_t)row * OUT_F + col] = f2bf(c[r] * av);
    hT_bf[(size_t)col * NROWS + row] = hb;
  }
}

// ---------------- Kernel 2: fused masked-softmax attention (flash) -----------
__global__ __launch_bounds__(128) void gat_flash(
    const int* __restrict__ adj,
    const unsigned short* __restrict__ h_bf, const unsigned short* __restrict__ g_bf,
    const unsigned short* __restrict__ hT_bf, float* __restrict__ out) {
  const int lane = threadIdx.x & 31;
  const int lm = lane & 15;
  const int hh = lane >> 4;
  const int waveId = blockIdx.x * (blockDim.x >> 5) + (threadIdx.x >> 5);
  const int rbase = waveId << 4;  // 16 attention rows per wave

  v8f acc[16];
#pragma unroll
  for (int t = 0; t < 16; ++t) acc[t] = 0.f;
  float rmax = NEG_BIG, rsum = 0.f;

  const unsigned short* gRow = g_bf + (size_t)(rbase + lm) * OUT_F + 16 * hh;

  for (int jb = 0; jb < NROWS; jb += 32) {
    // eT tiles: D[jloc][m] = sum_k h[jb+jloc,k] * g[rbase+m,k]
    v8f c0 = 0.f, c1 = 0.f;
    const unsigned short* hr0 = h_bf + (size_t)(jb + lm) * OUT_F + 16 * hh;
    const unsigned short* hr1 = hr0 + (size_t)16 * OUT_F;
#pragma unroll
    for (int k0 = 0; k0 < OUT_F; k0 += 32) {
      v16bf a0 = *(const v16bf*)(hr0 + k0);
      v16bf a1 = *(const v16bf*)(hr1 + k0);
      v16bf b  = *(const v16bf*)(gRow + k0);
      c0 = __builtin_amdgcn_wmma_f32_16x16x32_bf16(false, a0, false, b, (short)0, c0, false, false);
      c1 = __builtin_amdgcn_wmma_f32_16x16x32_bf16(false, a1, false, b, (short)0, c1, false, false);
    }
    // mask: lane holds row (rbase+lm), j = jb + r + 8*hh (c0) / +16 (c1)
    const int* arow = adj + (size_t)(rbase + lm) * NROWS + jb + 8 * hh;
    int4 m0 = *(const int4*)(arow);
    int4 m1 = *(const int4*)(arow + 4);
    int4 m2 = *(const int4*)(arow + 16);
    int4 m3 = *(const int4*)(arow + 20);
    int ma[8] = {m0.x, m0.y, m0.z, m0.w, m1.x, m1.y, m1.z, m1.w};
    int mb[8] = {m2.x, m2.y, m2.z, m2.w, m3.x, m3.y, m3.z, m3.w};
#pragma unroll
    for (int r = 0; r < 8; ++r) {
      c0[r] = (ma[r] > 0) ? c0[r] : NEG_BIG;
      c1[r] = (mb[r] > 0) ? c1[r] : NEG_BIG;
    }
    // online softmax (stats live per-lane: row = lm)
    float tmax = NEG_BIG;
#pragma unroll
    for (int r = 0; r < 8; ++r) tmax = fmaxf(tmax, fmaxf(c0[r], c1[r]));
    tmax = fmaxf(tmax, __shfl_xor(tmax, 16, 32));
    float nmax = fmaxf(rmax, tmax);
    float scale = __expf(rmax - nmax);
    float p0[8], p1[8], psum = 0.f;
#pragma unroll
    for (int r = 0; r < 8; ++r) {
      p0[r] = __expf(c0[r] - nmax);
      p1[r] = __expf(c1[r] - nmax);
      psum += p0[r] + p1[r];
    }
    psum += __shfl_xor(psum, 16, 32);
    rsum = rsum * scale + psum;
    rmax = nmax;
    // rescale accumulator rows (acc row index = r + 8*hh; stats at lane = row)
    float sc[8];
#pragma unroll
    for (int r = 0; r < 8; ++r) sc[r] = __shfl(scale, r + 8 * hh, 32);
#pragma unroll
    for (int t = 0; t < 16; ++t)
#pragma unroll
      for (int r = 0; r < 8; ++r) acc[t][r] *= sc[r];
    // repack P (16x32, row m = lm) into WMMA A fragment
    union { v16bf v; unsigned short s[16]; } pa;
#pragma unroll
    for (int r = 0; r < 8; ++r) {
      float q0 = __shfl_xor(p0[r], 16, 32);
      float q1 = __shfl_xor(p1[r], 16, 32);
      float lo = hh ? q1 : p0[r];   // elems 0..7
      float hi = hh ? p1[r] : q0;   // elems 8..15
      pa.s[r] = f2bf(lo);
      pa.s[8 + r] = f2bf(hi);
    }
    // O += P @ H_block, B from hT (contiguous per lane)
#pragma unroll
    for (int t = 0; t < 16; ++t) {
      v16bf bt = *(const v16bf*)(hT_bf + (size_t)(16 * t + lm) * NROWS + jb + 16 * hh);
      acc[t] = __builtin_amdgcn_wmma_f32_16x16x32_bf16(false, pa.v, false, bt, (short)0, acc[t], false, false);
    }
  }
  // finalize: divide by rowsum, elu, store
  float inv = 1.0f / rsum;
  float iv[8];
#pragma unroll
  for (int r = 0; r < 8; ++r) iv[r] = __shfl(inv, r + 8 * hh, 32);
#pragma unroll
  for (int t = 0; t < 16; ++t)
#pragma unroll
    for (int r = 0; r < 8; ++r) {
      float v = acc[t][r] * iv[r];
      v = (v > 0.f) ? v : (__expf(v) - 1.f);
      out[(size_t)(rbase + r + 8 * hh) * OUT_F + 16 * t + lm] = v;
    }
}

extern "C" void kernel_launch(void* const* d_in, const int* in_sizes, int n_in,
                              void* d_out, int out_size, void* d_ws, size_t ws_size,
                              hipStream_t stream) {
  const float* x   = (const float*)d_in[0];   // [8192, 512]
  const int*   adj = (const int*)d_in[1];     // [8192, 8192]
  const float* W   = (const float*)d_in[2];   // [512, 256]
  const float* a   = (const float*)d_in[3];   // [256, 1]
  float* out = (float*)d_out;                 // [8192, 256]

  char* ws = (char*)d_ws;
  unsigned short* x_bf  = (unsigned short*)ws;                       // 8 MB
  unsigned short* wt_bf = (unsigned short*)(ws + (size_t)8 * 1024 * 1024);   // 256 KB
  unsigned short* h_bf  = (unsigned short*)(ws + (size_t)8 * 1024 * 1024 + 262144);
  unsigned short* g_bf  = h_bf + (size_t)NROWS * OUT_F;              // +4 MB
  unsigned short* hT_bf = g_bf + (size_t)NROWS * OUT_F;              // +4 MB

  // x: 8192*512 = 4194304 elems -> 16384 blocks of 256
  gat_convert<<<16384, 256, 0, stream>>>(x, W, x_bf, wt_bf);
  // 512*16 = 8192 tiles, 4 waves/block -> 2048 blocks
  gat_hgemm<<<2048, 128, 0, stream>>>(x_bf, wt_bf, a, h_bf, g_bf, hT_bf);
  // 512 row-blocks, 4 waves/block -> 128 blocks
  gat_flash<<<128, 128, 0, stream>>>(adj, h_bf, g_bf, hT_bf, out);
}
